// LLcheM_69045894250506
// MI455X (gfx1250) — matrix operand order repaired and checked by reading, not compiled, not run on hardware
//

#include <hip/hip_runtime.h>
#include <hip/hip_bf16.h>

// ---------------------------------------------------------------------------
// Transformer encoder (S=1000, B=4, D=1024, H=16, L=4, FF=4096) for gfx1250.
// Strategy: residual stream in fp32; all GEMMs in bf16 via
// v_wmma_f32_16x16x32_bf16 (fp32 accumulate); flash attention with WMMA for
// QK^T and P*V, online softmax, LDS-based C->A fragment transpose.
// GEMM: each wave computes a 16x64 strip (4 accumulators); fragments are
// double-buffered across K-steps so loads overlap WMMA issue.
// ---------------------------------------------------------------------------

#define S_     1000
#define B_     4
#define D_     1024
#define H_     16
#define HD_    64
#define FF_    4096
#define L_     4
#define OUT_   64
#define SBROWS (S_ * B_)   // 4000 rows, divisible by 32
#define SPAD   1024        // padded key length for transposed V

typedef __attribute__((ext_vector_type(8)))  float   v8f;
typedef __attribute__((ext_vector_type(16))) __bf16  bf16x16;
typedef __attribute__((ext_vector_type(8)))  __bf16  bf16x8;
typedef __attribute__((ext_vector_type(4)))  __bf16  bf16x4;

struct BfPair { bf16x8 lo, hi; };
union Frag { bf16x16 v; BfPair p; };

__device__ __forceinline__ v8f wmma_bf16(bf16x16 a, bf16x16 b, v8f c) {
  // D = A(16x32) * B(32x16) + C, fp32 accumulate
  return __builtin_amdgcn_wmma_f32_16x16x32_bf16(
      /*neg_a=*/false, a, /*neg_b=*/false, b,
      /*c_mod=*/(short)0, c, /*reuse_a=*/false, /*reuse_b=*/false);
}

// ------------------------------ elementwise --------------------------------

__global__ void cast_f2b(const float* __restrict__ x, __bf16* __restrict__ o) {
  const int i = blockIdx.x * blockDim.x + threadIdx.x;
  o[i] = (__bf16)x[i];
}

__global__ void embed_g(const int* __restrict__ tok,
                        const float* __restrict__ emb,
                        float* __restrict__ x) {
  const int i  = blockIdx.x * blockDim.x + threadIdx.x;  // SBROWS*D
  const int d  = i & (D_ - 1);
  const int sb = i >> 10;
  x[i] = emb[(size_t)tok[sb] * D_ + d];
}

__global__ void gelu_cast(const float* __restrict__ x, __bf16* __restrict__ o) {
  const int i = blockIdx.x * blockDim.x + threadIdx.x;
  const float v = x[i];
  o[i] = (__bf16)(0.5f * v * (1.0f + erff(v * 0.70710678118654752f)));
}

// RoPE: x*cos + rotate_half(x)*sin, then cast to bf16 (layout [s,b,D] kept)
__global__ void rope_cast(const float* __restrict__ in, __bf16* __restrict__ out) {
  const int i  = blockIdx.x * blockDim.x + threadIdx.x;  // SBROWS*D
  const int d  = i & (D_ - 1);
  const int sb = i >> 10;
  const int s  = sb >> 2;                 // / B_
  const int p  = d & (HD_ - 1);
  const int pi = p & 31;
  const float freq = __powf(10000.0f, -(float)(2 * pi) * (1.0f / (float)HD_));
  const float ang  = (float)s * freq;
  float sn, cs;
  __sincosf(ang, &sn, &cs);
  const float xv  = in[i];
  const float xp  = in[i + ((p < 32) ? 32 : -32)];
  const float rot = (p < 32) ? -xp : xp;
  out[i] = (__bf16)(xv * cs + rot * sn);
}

// V: [s,b,D] fp32 -> Vt: [bh, hd, s(padded)] bf16, pad region zeroed
__global__ void vtrans_cast(const float* __restrict__ v, __bf16* __restrict__ vt) {
  const int i  = blockIdx.x * blockDim.x + threadIdx.x;  // B*H*HD*SPAD
  const int s  = i & (SPAD - 1);
  const int r  = i >> 10;        // bh*64 + hd
  const int hd = r & 63;
  const int bh = r >> 6;
  const int b  = bh >> 4;
  const int h  = bh & 15;
  float val = 0.0f;
  if (s < S_) val = v[((size_t)s * B_ + b) * D_ + h * HD_ + hd];
  vt[i] = (__bf16)val;
}

// ------------------------------- layernorm ---------------------------------

__global__ __launch_bounds__(256)
void ln_kernel(const float* __restrict__ x, const float* __restrict__ g,
               const float* __restrict__ bta, __bf16* __restrict__ out) {
  const int row = blockIdx.x;
  const float4 v = ((const float4*)(x + (size_t)row * D_))[threadIdx.x];
  __shared__ float red[8];

  float s = v.x + v.y + v.z + v.w;
  for (int off = 16; off > 0; off >>= 1) s += __shfl_xor(s, off, 32);
  if ((threadIdx.x & 31) == 0) red[threadIdx.x >> 5] = s;
  __syncthreads();
  float tot = 0.0f;
  #pragma unroll
  for (int i = 0; i < 8; ++i) tot += red[i];
  const float mean = tot * (1.0f / (float)D_);
  __syncthreads();

  const float d0 = v.x - mean, d1 = v.y - mean, d2 = v.z - mean, d3 = v.w - mean;
  float q = d0 * d0 + d1 * d1 + d2 * d2 + d3 * d3;
  for (int off = 16; off > 0; off >>= 1) q += __shfl_xor(q, off, 32);
  if ((threadIdx.x & 31) == 0) red[threadIdx.x >> 5] = q;
  __syncthreads();
  float var = 0.0f;
  #pragma unroll
  for (int i = 0; i < 8; ++i) var += red[i];
  const float inv = rsqrtf(var * (1.0f / (float)D_) + 1e-5f);

  const float4 gg = ((const float4*)g)[threadIdx.x];
  const float4 bb = ((const float4*)bta)[threadIdx.x];
  bf16x4 o;
  o[0] = (__bf16)(d0 * inv * gg.x + bb.x);
  o[1] = (__bf16)(d1 * inv * gg.y + bb.y);
  o[2] = (__bf16)(d2 * inv * gg.z + bb.z);
  o[3] = (__bf16)(d3 * inv * gg.w + bb.w);
  ((bf16x4*)(out + (size_t)row * D_))[threadIdx.x] = o;
}

// --------------------------------- GEMM ------------------------------------
// C[M,N] = alpha*(A[M,K] * W[N,K]^T + bias[N]) + resid[M,N]
// Block: 256 threads = 8 waves (2 along M x 4 along N) -> 32x256 tile.
// Per wave: 16x64 strip = 4 accumulators; one A fragment drives 4 WMMAs.
// K-loop is software-pipelined: step k+32 fragments are fetched before the
// step-k WMMAs are issued, so the loads overlap matrix-op execution.
// N not multiple of 256 handled by clamped B loads + guarded stores.

__global__ __launch_bounds__(256)
void gemm_bf16(const __bf16* __restrict__ A, const __bf16* __restrict__ W,
               const float* __restrict__ bias, const float* __restrict__ resid,
               float* __restrict__ C, int M, int N, int K, float alpha) {
  const int lane = threadIdx.x & 31;
  const int wave = threadIdx.x >> 5;
  const int l16  = lane & 15;
  const int lh   = lane >> 4;
  const int m0   = blockIdx.x * 32 + (wave & 1) * 16;
  const int nw   = blockIdx.y * 256 + (wave >> 1) * 64;   // wave's 64-col strip

  // A-frag: lane holds row m0+l16, 16 contiguous K at lh*16
  const __bf16* ap = A + (size_t)(m0 + l16) * K + lh * 16;

  // B-frags: lane holds column nw+g*16+l16, 16 contiguous K at lh*16
  const __bf16* bp[4];
  #pragma unroll
  for (int g = 0; g < 4; ++g) {
    int col = nw + g * 16 + l16;
    if (col >= N) col = N - 1;          // clamp; stores are guarded below
    bp[g] = W + (size_t)col * K + lh * 16;
  }

  // prologue: fragments for k = 0
  Frag a0;
  bf16x16 bb0[4];
  a0.p.lo = *(const bf16x8*)(ap);
  a0.p.hi = *(const bf16x8*)(ap + 8);
  #pragma unroll
  for (int g = 0; g < 4; ++g) bb0[g] = *(const bf16x16*)(bp[g]);

  v8f acc[4] = {{}, {}, {}, {}};
  #pragma unroll 2
  for (int kk = 32; kk < K; kk += 32) {
    // fetch next k-step before consuming the current one
    Frag a1;
    bf16x16 bb1[4];
    a1.p.lo = *(const bf16x8*)(ap + kk);
    a1.p.hi = *(const bf16x8*)(ap + kk + 8);
    #pragma unroll
    for (int g = 0; g < 4; ++g) bb1[g] = *(const bf16x16*)(bp[g] + kk);

    acc[0] = wmma_bf16(a0.v, bb0[0], acc[0]);
    acc[1] = wmma_bf16(a0.v, bb0[1], acc[1]);
    acc[2] = wmma_bf16(a0.v, bb0[2], acc[2]);
    acc[3] = wmma_bf16(a0.v, bb0[3], acc[3]);

    a0 = a1;
    #pragma unroll
    for (int g = 0; g < 4; ++g) bb0[g] = bb1[g];
  }
  // epilogue k-step
  acc[0] = wmma_bf16(a0.v, bb0[0], acc[0]);
  acc[1] = wmma_bf16(a0.v, bb0[1], acc[1]);
  acc[2] = wmma_bf16(a0.v, bb0[2], acc[2]);
  acc[3] = wmma_bf16(a0.v, bb0[3], acc[3]);

  #pragma unroll
  for (int g = 0; g < 4; ++g) {
    const int col = nw + g * 16 + l16;
    if (col < N) {
      const float bv = bias ? bias[col] : 0.0f;
      #pragma unroll
      for (int j = 0; j < 8; ++j) {
        const int row    = m0 + lh * 8 + j;  // C layout: VGPR j -> row j + 8*(lane/16)
        const size_t idx = (size_t)row * N + col;
        float v = alpha * (acc[g][j] + bv);
        if (resid) v += resid[idx];
        C[idx] = v;
      }
    }
  }
}

// ---------------------------- flash attention ------------------------------
// One wave handles 16 query rows for one (b,h). Key tiles of 32.
// Q,K bf16 in [s,b,D]; Vt bf16 in [bh,hd,SPAD]; O written bf16 to [s,b,D].

__global__ __launch_bounds__(256)
void attn_flash(const __bf16* __restrict__ qb, const __bf16* __restrict__ kb,
                const __bf16* __restrict__ vt, __bf16* __restrict__ ob) {
  __shared__ __bf16 plds[8][16][32];   // per-wave P transpose buffer (1 KB each)
  const int lane = threadIdx.x & 31;
  const int wave = threadIdx.x >> 5;
  const int l16  = lane & 15;
  const int lh   = lane >> 4;
  const int qt   = blockIdx.x * 8 + wave;
  const int m0   = qt * 16;
  if (m0 >= S_) return;                 // wave-uniform exit
  const int bh = blockIdx.y;
  const int b  = bh >> 4;
  const int h  = bh & 15;

  // Q A-fragments for hd [0,32) and [32,64)
  int qr = m0 + l16; if (qr >= S_) qr = S_ - 1;
  const __bf16* qp = qb + ((size_t)qr * B_ + b) * D_ + h * HD_;
  Frag qa0, qa1;
  qa0.p.lo = *(const bf16x8*)(qp + lh * 16);
  qa0.p.hi = *(const bf16x8*)(qp + lh * 16 + 8);
  qa1.p.lo = *(const bf16x8*)(qp + 32 + lh * 16);
  qa1.p.hi = *(const bf16x8*)(qp + 32 + lh * 16 + 8);

  float mrow[8], lrow[8];
  v8f o0 = {}, o1 = {}, o2 = {}, o3 = {};
  #pragma unroll
  for (int j = 0; j < 8; ++j) { mrow[j] = -1e30f; lrow[j] = 0.0f; }

  __bf16* pl = &plds[wave][0][0];
  const __bf16* vbase = vt + ((size_t)bh * HD_) * SPAD;
  const int NT = (S_ + 31) / 32;

  for (int t = 0; t < NT; ++t) {
    const int t0 = t * 32;
    // K B-fragments: lane = key column, 16 contiguous hd values at lh*16
    int kr0 = t0 + l16;       const bool kv0 = kr0 < S_; if (!kv0) kr0 = S_ - 1;
    int kr1 = t0 + 16 + l16;  const bool kv1 = kr1 < S_; if (!kv1) kr1 = S_ - 1;
    const __bf16* kp0 = kb + ((size_t)kr0 * B_ + b) * D_ + h * HD_ + lh * 16;
    const __bf16* kp1 = kb + ((size_t)kr1 * B_ + b) * D_ + h * HD_ + lh * 16;
    const bf16x16 b00 = *(const bf16x16*)(kp0);
    const bf16x16 b01 = *(const bf16x16*)(kp0 + 32);
    const bf16x16 b10 = *(const bf16x16*)(kp1);
    const bf16x16 b11 = *(const bf16x16*)(kp1 + 32);

    v8f s0 = {}, s1 = {};
    s0 = wmma_bf16(qa0.v, b00, s0);
    s0 = wmma_bf16(qa1.v, b01, s0);
    s1 = wmma_bf16(qa0.v, b10, s1);
    s1 = wmma_bf16(qa1.v, b11, s1);

    if (!kv0) {
      #pragma unroll
      for (int j = 0; j < 8; ++j) s0[j] = -1e30f;
    }
    if (!kv1) {
      #pragma unroll
      for (int j = 0; j < 8; ++j) s1[j] = -1e30f;
    }

    // row max across the 16 lanes holding each row (xor within half-wave)
    float nm[8];
    #pragma unroll
    for (int j = 0; j < 8; ++j) nm[j] = fmaxf(mrow[j], fmaxf(s0[j], s1[j]));
    #pragma unroll
    for (int off = 1; off <= 8; off <<= 1) {
      #pragma unroll
      for (int j = 0; j < 8; ++j) nm[j] = fmaxf(nm[j], __shfl_xor(nm[j], off, 32));
    }

    float pr0[8], pr1[8], rs[8];
    #pragma unroll
    for (int j = 0; j < 8; ++j) {
      const float f = __expf(mrow[j] - nm[j]);
      mrow[j] = nm[j];
      lrow[j] *= f;
      o0[j] *= f; o1[j] *= f; o2[j] *= f; o3[j] *= f;
      pr0[j] = kv0 ? __expf(s0[j] - nm[j]) : 0.0f;
      pr1[j] = kv1 ? __expf(s1[j] - nm[j]) : 0.0f;
      rs[j]  = pr0[j] + pr1[j];
    }
    #pragma unroll
    for (int off = 1; off <= 8; off <<= 1) {
      #pragma unroll
      for (int j = 0; j < 8; ++j) rs[j] += __shfl_xor(rs[j], off, 32);
    }
    #pragma unroll
    for (int j = 0; j < 8; ++j) lrow[j] += rs[j];

    // P: C layout -> LDS -> A-fragment layout
    #pragma unroll
    for (int j = 0; j < 8; ++j) {
      const int r = lh * 8 + j;
      pl[r * 32 + l16]      = (__bf16)pr0[j];
      pl[r * 32 + 16 + l16] = (__bf16)pr1[j];
    }
    asm volatile("s_wait_dscnt 0x0" ::: "memory");
    Frag pa;
    const __bf16* pp = pl + l16 * 32 + lh * 16;
    pa.p.lo = *(const bf16x8*)(pp);
    pa.p.hi = *(const bf16x8*)(pp + 8);

    // O += P(16x32) * V(32x64): Vt rows are hd, contiguous along s
    const bf16x16 vb0 = *(const bf16x16*)(vbase + (size_t)(0  + l16) * SPAD + t0 + lh * 16);
    const bf16x16 vb1 = *(const bf16x16*)(vbase + (size_t)(16 + l16) * SPAD + t0 + lh * 16);
    const bf16x16 vb2 = *(const bf16x16*)(vbase + (size_t)(32 + l16) * SPAD + t0 + lh * 16);
    const bf16x16 vb3 = *(const bf16x16*)(vbase + (size_t)(48 + l16) * SPAD + t0 + lh * 16);
    o0 = wmma_bf16(pa.v, vb0, o0);
    o1 = wmma_bf16(pa.v, vb1, o1);
    o2 = wmma_bf16(pa.v, vb2, o2);
    o3 = wmma_bf16(pa.v, vb3, o3);
  }

  #pragma unroll
  for (int j = 0; j < 8; ++j) {
    const int s = m0 + lh * 8 + j;
    if (s < S_) {
      const float inv = 1.0f / lrow[j];
      __bf16* op = ob + ((size_t)s * B_ + b) * D_ + h * HD_ + l16;
      op[0]  = (__bf16)(o0[j] * inv);
      op[16] = (__bf16)(o1[j] * inv);
      op[32] = (__bf16)(o2[j] * inv);
      op[48] = (__bf16)(o3[j] * inv);
    }
  }
}

// --------------------------------- host ------------------------------------

extern "C" void kernel_launch(void* const* d_in, const int* in_sizes, int n_in,
                              void* d_out, int out_size, void* d_ws, size_t ws_size,
                              hipStream_t stream) {
  (void)in_sizes; (void)n_in; (void)out_size; (void)ws_size;
  const int*   tokens = (const int*)d_in[0];
  const float* embedw = (const float*)d_in[1];
  const float* Wq = (const float*)d_in[2];   const float* bq = (const float*)d_in[3];
  const float* Wk = (const float*)d_in[4];   const float* bk = (const float*)d_in[5];
  const float* Wv = (const float*)d_in[6];   const float* bv = (const float*)d_in[7];
  const float* Wo = (const float*)d_in[8];   const float* bo = (const float*)d_in[9];
  const float* ln1g = (const float*)d_in[10]; const float* ln1b = (const float*)d_in[11];
  const float* ln2g = (const float*)d_in[12]; const float* ln2b = (const float*)d_in[13];
  const float* W1 = (const float*)d_in[14];  const float* b1 = (const float*)d_in[15];
  const float* W2 = (const float*)d_in[16];  const float* b2 = (const float*)d_in[17];
  const float* Wout = (const float*)d_in[18]; const float* bout = (const float*)d_in[19];

  char* ws = (char*)d_ws;
  const size_t MB = 1ull << 20;
  __bf16* wq_b   = (__bf16*)(ws);             //  8 MB
  __bf16* wk_b   = (__bf16*)(ws + 8 * MB);    //  8 MB
  __bf16* wv_b   = (__bf16*)(ws + 16 * MB);   //  8 MB
  __bf16* wo_b   = (__bf16*)(ws + 24 * MB);   //  8 MB
  __bf16* w1_b   = (__bf16*)(ws + 32 * MB);   // 32 MB
  __bf16* w2_b   = (__bf16*)(ws + 64 * MB);   // 32 MB
  __bf16* wout_b = (__bf16*)(ws + 96 * MB);   // 128 KB (1 MB slot)
  float*  x      = (float*)(ws + 97 * MB);    // 16 MB residual stream
  __bf16* hb     = (__bf16*)(ws + 113 * MB);  //  8 MB LN/cast output
  float*  h1     = (float*)(ws + 121 * MB);   // 64 MB FFN hidden, aliases q/k/v
  __bf16* h1b    = (__bf16*)(ws + 185 * MB);  // 32 MB gelu out, aliases qb/kb/vt/ob

  float*  qf = h1;
  float*  kf = (float*)((char*)h1 + 16 * MB);
  float*  vf = (float*)((char*)h1 + 32 * MB);
  __bf16* qbuf = h1b;
  __bf16* kbuf = (__bf16*)((char*)h1b + 8 * MB);
  __bf16* vtb  = (__bf16*)((char*)h1b + 16 * MB);
  __bf16* obuf = (__bf16*)((char*)h1b + 24 * MB);

  const dim3 blk(256);
  // one-time (per launch) bf16 weight casts
  cast_f2b<<<(L_ * D_ * D_) / 256, blk, 0, stream>>>(Wq, wq_b);
  cast_f2b<<<(L_ * D_ * D_) / 256, blk, 0, stream>>>(Wk, wk_b);
  cast_f2b<<<(L_ * D_ * D_) / 256, blk, 0, stream>>>(Wv, wv_b);
  cast_f2b<<<(L_ * D_ * D_) / 256, blk, 0, stream>>>(Wo, wo_b);
  cast_f2b<<<(L_ * FF_ * D_) / 256, blk, 0, stream>>>(W1, w1_b);
  cast_f2b<<<(L_ * D_ * FF_) / 256, blk, 0, stream>>>(W2, w2_b);
  cast_f2b<<<(OUT_ * D_) / 256, blk, 0, stream>>>(Wout, wout_b);

  embed_g<<<(SBROWS * D_) / 256, blk, 0, stream>>>(tokens, embedw, x);

  const float scale = 0.125f;  // HD^-0.5
  const dim3 gD(SBROWS / 32, D_ / 256);     // N = 1024
  const dim3 gFF(SBROWS / 32, FF_ / 256);   // N = 4096
  const dim3 gOUT(SBROWS / 32, 1);          // N = 64 (guarded)

  for (int i = 0; i < L_; ++i) {
    ln_kernel<<<SBROWS, blk, 0, stream>>>(x, ln1g + i * D_, ln1b + i * D_, hb);
    gemm_bf16<<<gD, blk, 0, stream>>>(hb, wq_b + (size_t)i * D_ * D_, bq + i * D_,
                                      nullptr, qf, SBROWS, D_, D_, scale);
    gemm_bf16<<<gD, blk, 0, stream>>>(hb, wk_b + (size_t)i * D_ * D_, bk + i * D_,
                                      nullptr, kf, SBROWS, D_, D_, 1.0f);
    gemm_bf16<<<gD, blk, 0, stream>>>(hb, wv_b + (size_t)i * D_ * D_, bv + i * D_,
                                      nullptr, vf, SBROWS, D_, D_, 1.0f);
    rope_cast<<<(SBROWS * D_) / 256, blk, 0, stream>>>(qf, qbuf);
    rope_cast<<<(SBROWS * D_) / 256, blk, 0, stream>>>(kf, kbuf);
    vtrans_cast<<<(B_ * H_ * HD_ * SPAD) / 256, blk, 0, stream>>>(vf, vtb);
    attn_flash<<<dim3(8, B_ * H_), blk, 0, stream>>>(qbuf, kbuf, vtb, obuf);
    gemm_bf16<<<gD, blk, 0, stream>>>(obuf, wo_b + (size_t)i * D_ * D_, bo + i * D_,
                                      x, x, SBROWS, D_, D_, 1.0f);
    ln_kernel<<<SBROWS, blk, 0, stream>>>(x, ln2g + i * D_, ln2b + i * D_, hb);
    gemm_bf16<<<gFF, blk, 0, stream>>>(hb, w1_b + (size_t)i * FF_ * D_, b1 + i * FF_,
                                       nullptr, h1, SBROWS, FF_, D_, 1.0f);
    gelu_cast<<<(SBROWS * FF_) / 256, blk, 0, stream>>>(h1, h1b);
    gemm_bf16<<<gD, blk, 0, stream>>>(h1b, w2_b + (size_t)i * D_ * FF_, b2 + i * D_,
                                      x, x, SBROWS, D_, FF_, 1.0f);
  }

  cast_f2b<<<(SBROWS * D_) / 256, blk, 0, stream>>>(x, hb);
  gemm_bf16<<<gOUT, blk, 0, stream>>>(hb, wout_b, bout, nullptr, (float*)d_out,
                                      SBROWS, OUT_, D_, 1.0f);
}
